// CausalSelfAttention_87849261072913
// MI455X (gfx1250) — compile-verified, matrix operand order checked by reading, not run on hardware
//
#include <hip/hip_runtime.h>
#include <hip/hip_bf16.h>
#include <stdint.h>

typedef __attribute__((ext_vector_type(16))) _Float16 v16h;
typedef __attribute__((ext_vector_type(8)))  _Float16 v8h;
typedef __attribute__((ext_vector_type(8)))  float    v8f;

constexpr int B_   = 2;
constexpr int T_   = 2048;
constexpr int C_   = 1024;
constexpr int H_   = 16;
constexpr int HKV_ = 4;
constexpr int HD_  = 64;
constexpr int G_   = 4;       // H / HKV
constexpr int M_   = B_ * T_; // 4096 rows of x
constexpr int NKV_ = HKV_ * HD_; // 256

// Load a 16-bit WMMA fragment slice for one lane.
// Per CDNA5 ISA 7.12.2 (16-bit A / B layouts): elements 0..7 sit at p[0..7],
// elements 8..15 at p[16..23]  (the +8 K-offset comes from the caller via hb).
__device__ __forceinline__ v16h load_frag(const _Float16* p) {
  v8h lo = *(const v8h*)(p);
  v8h hi = *(const v8h*)(p + 16);
  return __builtin_shufflevector(lo, hi, 0,1,2,3,4,5,6,7,8,9,10,11,12,13,14,15);
}

__global__ void cvt_f16_kernel(const float* __restrict__ in, _Float16* __restrict__ out, int n) {
  int i = blockIdx.x * blockDim.x + threadIdx.x;
  if (i < n) out[i] = (_Float16)in[i];
}

// W: [K][N] f32  ->  Wt: [N][K] f16   (coalesced writes)
__global__ void transpose_cvt_kernel(const float* __restrict__ W, _Float16* __restrict__ Wt,
                                     int K, int N) {
  int i = blockIdx.x * blockDim.x + threadIdx.x;
  if (i < K * N) {
    int k = i % K, n = i / K;
    Wt[i] = (_Float16)W[(size_t)k * N + n];
  }
}

// C[M][N] = A[M][K] * Bt[N][K]^T, f16 in, f32 accumulate via v_wmma_f32_16x16x32_f16.
// Block = 128 threads (4 waves) -> 64x64 macro-tile; wave w owns rows w*16..w*16+15.
// The shared 64x32 B tile per K-step is DMA'd into LDS with
// global_load_async_to_lds_b128 (ASYNCcnt), double-buffered to overlap with WMMA.
// OUTMODE 0: f32 plain [M][N]   (final output)
// OUTMODE 1: f16 plain [M][N]   (Q, K buffers)
// OUTMODE 2: f16 scatter to V-transposed [b][kv][hd][t]  (V buffer)
template<int OUTMODE>
__global__ void gemm_wmma_kernel(const _Float16* __restrict__ A, const _Float16* __restrict__ Bt,
                                 void* __restrict__ Cv, int Md, int Nd, int Kd) {
  __shared__ __align__(16) _Float16 ldsB[2][64 * 32];   // 2 x 4 KB double buffer
  int tid  = threadIdx.x;
  int lane = tid & 31;
  int wave = tid >> 5;
  int l16  = lane & 15;
  int hb   = (lane >> 4) * 8;          // K-base 0 / 8 per half-wave
  int m64tiles = Md >> 6;
  int m64 = blockIdx.x % m64tiles;
  int n64 = blockIdx.x / m64tiles;

  const _Float16* bbase = Bt + (size_t)(n64 * 64) * Kd;
  const _Float16* arow  = A + (size_t)(m64 * 64 + wave * 16 + l16) * Kd;

  v8f acc[4] = {v8f{}, v8f{}, v8f{}, v8f{}};

  // Stage the 64(col) x 32(k) f16 B tile (4 KB) into ldsB[buf]:
  // 256 chunks of 16 B, 128 lanes x 2 async ops.
  auto stage = [&](int buf, int k0) {
    #pragma unroll
    for (int i = 0; i < 2; ++i) {
      int c   = tid + i * 128;
      int col = c >> 2;            // 0..63
      int ko  = (c & 3) * 8;       // 0,8,16,24 halves
      unsigned long long g =
          (unsigned long long)(uintptr_t)(bbase + (size_t)col * Kd + k0 + ko);
      unsigned l = (unsigned)(uintptr_t)(&ldsB[buf][col * 32 + ko]);
      asm volatile("global_load_async_to_lds_b128 %0, %1, off"
                   :: "v"(l), "v"(g) : "memory");
    }
  };

  int buf = 0;
  stage(0, 0);
  for (int k0 = 0; k0 < Kd; k0 += 32) {
    asm volatile("s_wait_asynccnt 0x0" ::: "memory");  // current buf resident
    __syncthreads();
    if (k0 + 32 < Kd) stage(buf ^ 1, k0 + 32);         // overlap next DMA with math
    if (k0 + 128 < Kd) __builtin_prefetch(arow + k0 + 128);
    v16h af = load_frag(arow + k0 + hb);               // per-wave A rows from global
    #pragma unroll
    for (int t = 0; t < 4; ++t) {
      v16h bf = load_frag(&ldsB[buf][(t * 16 + l16) * 32 + hb]);   // ds_load_b128 x2
      acc[t] = __builtin_amdgcn_wmma_f32_16x16x32_f16(false, af, false, bf,
                                                      (short)0, acc[t], false, false);
    }
    __syncthreads();                                   // reads done before buf reuse
    buf ^= 1;
  }

  #pragma unroll
  for (int t = 0; t < 4; ++t) {
    #pragma unroll
    for (int r = 0; r < 8; ++r) {
      int row = m64 * 64 + wave * 16 + r + hb;  // D layout: M = r + 8*(lane>=16)
      int col = n64 * 64 + t * 16 + l16;        // N = lane%16
      float v = acc[t][r];
      if (OUTMODE == 0) {
        ((float*)Cv)[(size_t)row * Nd + col] = v;
      } else if (OUTMODE == 1) {
        ((_Float16*)Cv)[(size_t)row * Nd + col] = (_Float16)v;
      } else {
        int b = row / T_, tt = row % T_;
        int kv = col / HD_, hd = col % HD_;
        ((_Float16*)Cv)[((size_t)((b * HKV_ + kv) * HD_ + hd)) * T_ + tt] = (_Float16)v;
      }
    }
  }
}

// Flash attention: 1 wave per (b, h, 16-row query tile). Online softmax,
// causal + ALiBi, key tiles of 32; S and PV both via v_wmma_f32_16x16x32_f16.
__global__ void attn_kernel(const _Float16* __restrict__ Qh, const _Float16* __restrict__ Kh,
                            const _Float16* __restrict__ Vt, _Float16* __restrict__ Ah) {
  __shared__ __align__(32) float lds[16 * 32];  // 16 q-rows x 32 key-cols of P
  int lane = threadIdx.x;
  int l16  = lane & 15;
  int hb   = (lane >> 4) * 8;
  int id   = blockIdx.x;
  int qt   = id % (T_ / 16);
  int h    = (id / (T_ / 16)) % H_;
  int b    = id / ((T_ / 16) * H_);
  int kv   = h / G_;
  float slope = exp2f(-0.5f * (float)(kv + 1));  // alibi_slopes(16)[:4][kv]
  int t0 = qt * 16;

  const _Float16* qbase = Qh + ((size_t)(b * T_ + t0 + l16)) * C_ + h * HD_;
  v16h qf0 = load_frag(qbase + hb);        // hd 0..31 contraction slice
  v16h qf1 = load_frag(qbase + 32 + hb);   // hd 32..63

  v8f o[4] = {v8f{}, v8f{}, v8f{}, v8f{}};
  float m[8], s[8];
  #pragma unroll
  for (int r = 0; r < 8; ++r) { m[r] = -__builtin_inff(); s[r] = 0.f; }

  const _Float16* kbase = Kh + ((size_t)b * T_) * NKV_ + kv * HD_;
  const _Float16* vbase = Vt + ((size_t)(b * HKV_ + kv)) * HD_ * T_;

  int qhi = t0 + 15;
  for (int j0 = 0; j0 <= qhi; j0 += 32) {
    // ---- S = Q * K^T for keys [j0, j0+32) : two 16x16 D tiles ----
    v8f sc[2];
    #pragma unroll
    for (int nt = 0; nt < 2; ++nt) {
      int key = j0 + nt * 16 + l16;                 // B layout: N = lane%16
      const _Float16* kp = kbase + (size_t)key * NKV_;
      v16h kf0 = load_frag(kp + hb);
      v16h kf1 = load_frag(kp + 32 + hb);
      v8f c = {};
      c = __builtin_amdgcn_wmma_f32_16x16x32_f16(false, qf0, false, kf0, (short)0, c, false, false);
      c = __builtin_amdgcn_wmma_f32_16x16x32_f16(false, qf1, false, kf1, (short)0, c, false, false);
      sc[nt] = c;
    }
    // ---- scale + ALiBi + causal mask, row max ----
    float vals[2][8], rmax[8];
    #pragma unroll
    for (int r = 0; r < 8; ++r) rmax[r] = -__builtin_inff();
    #pragma unroll
    for (int nt = 0; nt < 2; ++nt) {
      int j = j0 + nt * 16 + l16;
      #pragma unroll
      for (int r = 0; r < 8; ++r) {
        int i = t0 + r + hb;
        float v = sc[nt][r] * 0.125f + slope * (float)(j - i);
        if (j > i) v = -__builtin_inff();
        vals[nt][r] = v;
        rmax[r] = fmaxf(rmax[r], v);
      }
    }
    #pragma unroll
    for (int r = 0; r < 8; ++r) {  // reduce over the 16-lane N group
      float v = rmax[r];
      v = fmaxf(v, __shfl_xor(v, 1, 32));
      v = fmaxf(v, __shfl_xor(v, 2, 32));
      v = fmaxf(v, __shfl_xor(v, 4, 32));
      v = fmaxf(v, __shfl_xor(v, 8, 32));
      rmax[r] = v;
    }
    float corr[8];
    #pragma unroll
    for (int r = 0; r < 8; ++r) {
      float mn = fmaxf(m[r], rmax[r]);
      corr[r] = expf(m[r] - mn);   // expf(-inf)=0 on first tile
      m[r] = mn;
    }
    // ---- P = exp(S - m); stage D-layout -> LDS for A-layout transpose ----
    #pragma unroll
    for (int nt = 0; nt < 2; ++nt) {
      #pragma unroll
      for (int r = 0; r < 8; ++r) {
        float p = expf(vals[nt][r] - m[r]);
        vals[nt][r] = p;
        lds[(r + hb) * 32 + nt * 16 + l16] = p;
      }
    }
    #pragma unroll
    for (int r = 0; r < 8; ++r) {
      float ps = vals[0][r] + vals[1][r];
      ps += __shfl_xor(ps, 1, 32);
      ps += __shfl_xor(ps, 2, 32);
      ps += __shfl_xor(ps, 4, 32);
      ps += __shfl_xor(ps, 8, 32);
      s[r] = s[r] * corr[r] + ps;
      #pragma unroll
      for (int t = 0; t < 4; ++t) o[t][r] *= corr[r];   // rescale running O
    }
    asm volatile("s_wait_dscnt 0x0" ::: "memory");       // LDS RAW fence (single wave)
    const float* lp = lds + l16 * 32 + hb;               // A layout: M=lane%16
    v8f plo = *(const v8f*)(lp);
    v8f phi = *(const v8f*)(lp + 16);
    v16h pf;
    #pragma unroll
    for (int e = 0; e < 8; ++e) { pf[e] = (_Float16)plo[e]; pf[8 + e] = (_Float16)phi[e]; }
    // ---- O += P * V : 4 N-tiles over HD=64, contraction = 32 keys ----
    #pragma unroll
    for (int ht = 0; ht < 4; ++ht) {
      int hd = ht * 16 + l16;
      v16h vf = load_frag(vbase + (size_t)hd * T_ + j0 + hb);  // V^T rows contiguous
      o[ht] = __builtin_amdgcn_wmma_f32_16x16x32_f16(false, pf, false, vf,
                                                     (short)0, o[ht], false, false);
    }
  }
  // ---- normalize + store merged-head layout [b][t][h*64+hd] as f16 ----
  #pragma unroll
  for (int ht = 0; ht < 4; ++ht) {
    #pragma unroll
    for (int r = 0; r < 8; ++r) {
      int trow = t0 + r + hb;
      int col  = h * HD_ + ht * 16 + l16;
      Ah[((size_t)(b * T_ + trow)) * C_ + col] = (_Float16)(o[ht][r] / s[r]);
    }
  }
}

extern "C" void kernel_launch(void* const* d_in, const int* in_sizes, int n_in,
                              void* d_out, int out_size, void* d_ws, size_t ws_size,
                              hipStream_t stream) {
  const float* x  = (const float*)d_in[0];
  const float* Wq = (const float*)d_in[1];
  const float* Wk = (const float*)d_in[2];
  const float* Wv = (const float*)d_in[3];
  const float* Wo = (const float*)d_in[4];

  _Float16* ws = (_Float16*)d_ws;
  size_t off = 0;
  _Float16* Xh  = ws + off; off += (size_t)M_ * C_;     // x in f16
  _Float16* Qh  = ws + off; off += (size_t)M_ * C_;     // Q [b][t][h*64+hd]
  _Float16* Kh  = ws + off; off += (size_t)M_ * NKV_;   // K [b][t][kv*64+hd]
  _Float16* Vh  = ws + off; off += (size_t)M_ * NKV_;   // V^T [b][kv][hd][t]
  _Float16* WqT = ws + off; off += (size_t)C_ * C_;
  _Float16* WkT = ws + off; off += (size_t)NKV_ * C_;
  _Float16* WvT = ws + off; off += (size_t)NKV_ * C_;
  _Float16* WoT = ws + off; off += (size_t)C_ * C_;
  _Float16* Ah  = ws + off; off += (size_t)M_ * C_;     // attention output f16
  (void)ws_size; (void)in_sizes; (void)n_in; (void)out_size;

  cvt_f16_kernel<<<(M_ * C_ + 255) / 256, 256, 0, stream>>>(x, Xh, M_ * C_);
  transpose_cvt_kernel<<<(C_ * C_   + 255) / 256, 256, 0, stream>>>(Wq, WqT, C_, C_);
  transpose_cvt_kernel<<<(C_ * NKV_ + 255) / 256, 256, 0, stream>>>(Wk, WkT, C_, NKV_);
  transpose_cvt_kernel<<<(C_ * NKV_ + 255) / 256, 256, 0, stream>>>(Wv, WvT, C_, NKV_);
  transpose_cvt_kernel<<<(C_ * C_   + 255) / 256, 256, 0, stream>>>(Wo, WoT, C_, C_);

  gemm_wmma_kernel<1><<<(M_ / 64) * (C_   / 64), 128, 0, stream>>>(Xh, WqT, Qh, M_, C_,   C_);
  gemm_wmma_kernel<1><<<(M_ / 64) * (NKV_ / 64), 128, 0, stream>>>(Xh, WkT, Kh, M_, NKV_, C_);
  gemm_wmma_kernel<2><<<(M_ / 64) * (NKV_ / 64), 128, 0, stream>>>(Xh, WvT, Vh, M_, NKV_, C_);

  attn_kernel<<<B_ * H_ * (T_ / 16), 32, 0, stream>>>(Qh, Kh, Vh, Ah);

  gemm_wmma_kernel<0><<<(M_ / 64) * (C_ / 64), 128, 0, stream>>>(Ah, WoT, d_out, M_, C_, C_);
}
